// SigmoidGatedAttention_66322884985392
// MI455X (gfx1250) — matrix operand order repaired and checked by reading, not compiled, not run on hardware
//
#include <hip/hip_runtime.h>

// ---- problem constants (B=1) ----
#define TT   4096
#define DD   1024
#define HH   16
#define DKK  64
#define FF   4096   // 4*D

typedef __bf16 bf16_t;
typedef __attribute__((ext_vector_type(16))) __bf16 v16bf;
typedef __attribute__((ext_vector_type(8)))  __bf16 v8bf;
typedef __attribute__((ext_vector_type(8)))  float  v8f;

// ---------------------------------------------------------------------------
// A-fragment loader (16x32 bf16, M x K). contiguous k order used here (B reads match):
//   lane L: M = L%16 ; elem j -> K = (L/16)*16 + j
// ---------------------------------------------------------------------------
__device__ inline v16bf load_afrag(const bf16_t* __restrict__ rowptr, int k0, int sub) {
  v8bf lo = *(const v8bf*)(rowptr + k0 + sub * 16);
  v8bf hi = *(const v8bf*)(rowptr + k0 + sub * 16 + 8);
  v16bf a;
#pragma unroll
  for (int i = 0; i < 8; ++i) { a[i] = lo[i]; a[8 + i] = hi[i]; }
  return a;
}

__device__ inline v8f zero_v8f() {
  v8f z;
#pragma unroll
  for (int r = 0; r < 8; ++r) z[r] = 0.0f;
  return z;
}

// CDNA5 async global->LDS copy (ASYNCcnt-tracked), 16 bytes per lane.
// GV mode: lds[vdst_lane] = mem[vaddr_lane]   (ISA 08_async_tensor §4.4)
__device__ inline void async_copy_b128(uint32_t lds_off, const bf16_t* gsrc) {
  asm volatile("global_load_async_to_lds_b128 %0, %1, off"
               :: "v"(lds_off), "v"(gsrc)
               : "memory");
}
__device__ inline void wait_asynccnt0() {
  asm volatile("s_wait_asynccnt 0" ::: "memory");
}

// ---------------------------------------------------------------------------
// Weight convert + transpose via LDS tile (coalesced both sides):
//   Wt[n*K+k] = bf16(W[k*N+n]);  grid = (N/32, K/32), block = 256
// ---------------------------------------------------------------------------
__global__ __launch_bounds__(256)
void wconv_kernel(const float* __restrict__ W, bf16_t* __restrict__ Wt,
                  int K, int N) {
  __shared__ float tile[32][33];
  int n0 = blockIdx.x * 32;
  int k0 = blockIdx.y * 32;
  int tx = threadIdx.x & 31;
  int ty = threadIdx.x >> 5;  // 0..7
#pragma unroll
  for (int i = 0; i < 4; ++i)
    tile[ty + 8 * i][tx] = W[(size_t)(k0 + ty + 8 * i) * N + n0 + tx];
  __syncthreads();
#pragma unroll
  for (int i = 0; i < 4; ++i)
    Wt[(size_t)(n0 + ty + 8 * i) * K + k0 + tx] = (bf16_t)tile[tx][ty + 8 * i];
}

// ---------------------------------------------------------------------------
// LayerNorm over D=1024. in = a (+ b if non-null). Writes bf16 normalized
// output; optionally writes the (a+b) sum as fp32 (residual stream y).
// ---------------------------------------------------------------------------
__global__ __launch_bounds__(256)
void ln_kernel(const float* __restrict__ a, const float* __restrict__ b,
               const float* __restrict__ gamma, const float* __restrict__ beta,
               bf16_t* __restrict__ out_bf, float* __restrict__ y_out) {
  const int D = DD;
  int row = blockIdx.x;
  int tid = threadIdx.x;
  __shared__ float s1[256];
  __shared__ float s2[256];
  float l1 = 0.f, l2 = 0.f;
  for (int c = tid; c < D; c += 256) {
    float v = a[(size_t)row * D + c];
    if (b) v += b[(size_t)row * D + c];
    if (y_out) y_out[(size_t)row * D + c] = v;
    l1 += v;
    l2 += v * v;
  }
  s1[tid] = l1; s2[tid] = l2;
  __syncthreads();
  for (int o = 128; o > 0; o >>= 1) {
    if (tid < o) { s1[tid] += s1[tid + o]; s2[tid] += s2[tid + o]; }
    __syncthreads();
  }
  float mu  = s1[0] * (1.0f / D);
  float var = s2[0] * (1.0f / D) - mu * mu;
  float inv = rsqrtf(var + 1e-5f);
  for (int c = tid; c < D; c += 256) {
    float v = a[(size_t)row * D + c];
    if (b) v += b[(size_t)row * D + c];
    out_bf[(size_t)row * D + c] = (bf16_t)((v - mu) * inv * gamma[c] + beta[c]);
  }
}

// ---------------------------------------------------------------------------
// Tiled WMMA GEMM with async-LDS double-buffered B panel.
//   C[M][N] = A[M][K] * Bt[N][K]^T (+bias), fused epilogue.
//   block = 256 threads = 8 waves; block tile = 128M x 128N; K-step 32.
//   B panel (128 cols x 32 k = 8KB bf16) staged to LDS with
//   global_load_async_to_lds_b128, double buffered; prefetch of step i+1
//   overlaps the 8 WMMAs of step i. All 8 B fragments are loaded before the
//   WMMA burst so the DS waits are partial (in-order DS completion).
//   mode 0: Cf = acc + bias            (fp32 store)
//   mode 1: Cbf = gelu(acc + bias)     (bf16 store)
//   mode 2: Cf = acc + bias + extra    (fp32 store, residual)
// ---------------------------------------------------------------------------
__global__ __launch_bounds__(256)
void gemm_bf16_kernel(const bf16_t* __restrict__ A, const bf16_t* __restrict__ Bt,
                      const float* __restrict__ bias, float* __restrict__ Cf,
                      bf16_t* __restrict__ Cbf, const float* __restrict__ extra,
                      int M, int N, int K, int mode) {
  const int tid  = threadIdx.x;
  const int lane = tid & 31;
  const int wave = tid >> 5;
  const int col  = lane & 15;
  const int sub  = lane >> 4;
  const int m0 = blockIdx.y * 128 + wave * 16;
  const int n0 = blockIdx.x * 128;

  // B panel staging buffers: [col 0..127][k 0..31] bf16, 64B per col-row
  __shared__ __align__(64) bf16_t bsm[2][128 * 32];

  // stage: 512 chunks of 16B; each thread copies 2 chunks
  auto stage = [&](bf16_t* buf, int k0) {
#pragma unroll
    for (int i = 0; i < 2; ++i) {
      int c = tid + i * 256;          // 0..511
      int colIdx = c >> 2;            // 0..127
      int kc = c & 3;                 // 16B chunk within 64B row
      const bf16_t* src = Bt + (size_t)(n0 + colIdx) * K + k0 + kc * 8;
      uint32_t dst = (uint32_t)(uintptr_t)(buf + colIdx * 32 + kc * 8);
      async_copy_b128(dst, src);
    }
  };

  v8f acc[8];
#pragma unroll
  for (int j = 0; j < 8; ++j) acc[j] = zero_v8f();

  const bf16_t* arow = A + (size_t)(m0 + col) * K;

  stage(bsm[0], 0);
  int buf = 0;
  for (int k0 = 0; k0 < K; k0 += 32) {
    // data for this k-step staged last iteration (or just above):
    wait_asynccnt0();
    __syncthreads();
    if (k0 + 32 < K) stage(bsm[buf ^ 1], k0 + 32);

    v16bf a = load_afrag(arow, k0, sub);
    const bf16_t* bbase = bsm[buf] + col * 32 + sub * 16;  // + j*512 per tile
    v16bf bfrag[8];
#pragma unroll
    for (int j = 0; j < 8; ++j)
      bfrag[j] = *(const v16bf*)(bbase + j * 512);
#pragma unroll
    for (int j = 0; j < 8; ++j)
      acc[j] = __builtin_amdgcn_wmma_f32_16x16x32_bf16(
          false, a, false, bfrag[j], (short)0, acc[j], false, false);
    buf ^= 1;
  }

#pragma unroll
  for (int j = 0; j < 8; ++j) {
    int n = n0 + j * 16 + col;
    float bv = bias ? bias[n] : 0.f;
#pragma unroll
    for (int r = 0; r < 8; ++r) {
      int m = m0 + r + 8 * sub;
      size_t idx = (size_t)m * N + n;
      float v = acc[j][r] + bv;
      if (mode == 0) {
        Cf[idx] = v;
      } else if (mode == 1) {
        Cbf[idx] = (bf16_t)(0.5f * v * (1.0f + erff(v * 0.70710678118f)));
      } else {
        Cf[idx] = v + extra[idx];
      }
    }
  }
}

// ---------------------------------------------------------------------------
// RoPE + head-major relayout: raw[t][h*64+d] (fp32) -> out[h][t][d] (bf16)
// ---------------------------------------------------------------------------
__global__ void rope_kernel(const float* __restrict__ raw, const float* __restrict__ cosb,
                            const float* __restrict__ sinb, bf16_t* __restrict__ outhm) {
  int idx = blockIdx.x * blockDim.x + threadIdx.x;
  if (idx >= TT * HH * 32) return;
  int d = idx & 31;
  int h = (idx >> 5) & (HH - 1);
  int t = idx >> 9;
  float x1 = raw[(size_t)t * DD + h * DKK + d];
  float x2 = raw[(size_t)t * DD + h * DKK + 32 + d];
  float c = cosb[(size_t)t * DKK + d];
  float s = sinb[(size_t)t * DKK + d];
  size_t base = (size_t)h * TT * DKK + (size_t)t * DKK;
  outhm[base + d]      = (bf16_t)(x1 * c - x2 * s);
  outhm[base + 32 + d] = (bf16_t)(x1 * s + x2 * c);
}

// ---------------------------------------------------------------------------
// V relayout transposed via LDS tile: raw[t][h*64+d] -> vt[h][d][t] (bf16)
//   grid = (T/32, D/32), block = 256
// ---------------------------------------------------------------------------
__global__ __launch_bounds__(256)
void vtrans_kernel(const float* __restrict__ raw, bf16_t* __restrict__ vt) {
  __shared__ float tile[32][33];
  int t0 = blockIdx.x * 32;
  int c0 = blockIdx.y * 32;     // column within D; stays inside one head (64|32)
  int tx = threadIdx.x & 31;
  int ty = threadIdx.x >> 5;
#pragma unroll
  for (int i = 0; i < 4; ++i)
    tile[ty + 8 * i][tx] = raw[(size_t)(t0 + ty + 8 * i) * DD + c0 + tx];
  __syncthreads();
  int h = c0 >> 6;
  int dbase = c0 & 63;
#pragma unroll
  for (int i = 0; i < 4; ++i)
    vt[(size_t)h * DKK * TT + (size_t)(dbase + ty + 8 * i) * TT + t0 + tx] =
        (bf16_t)tile[tx][ty + 8 * i];
}

// ---------------------------------------------------------------------------
// Causal flash attention with sigmoid gate epilogue.
// One wave per (16-query tile, head). Q*K^T and P*V via WMMA bf16.
// P goes through LDS to convert C-layout -> A-layout.
// ---------------------------------------------------------------------------
__global__ __launch_bounds__(32)
void attn_kernel(const bf16_t* __restrict__ qbf, const bf16_t* __restrict__ kbf,
                 const bf16_t* __restrict__ vt, const float* __restrict__ gateraw,
                 bf16_t* __restrict__ gated) {
  const int lane = threadIdx.x & 31;
  const int col  = lane & 15;
  const int sub  = lane >> 4;
  const int m0   = blockIdx.x * 16;
  const int h    = blockIdx.y;
  const float scale = 0.125f;  // 1/sqrt(64)

  const bf16_t* Q  = qbf + (size_t)h * TT * DKK;
  const bf16_t* Kp = kbf + (size_t)h * TT * DKK;
  const bf16_t* Vp = vt  + (size_t)h * DKK * TT;

  __shared__ __align__(32) bf16_t pbuf[16 * 32];

  const bf16_t* qrow = Q + (size_t)(m0 + col) * DKK;
  v16bf qa0 = load_afrag(qrow, 0, sub);
  v16bf qa1 = load_afrag(qrow, 32, sub);

  v8f o[4];
#pragma unroll
  for (int dt = 0; dt < 4; ++dt) o[dt] = zero_v8f();
  float mr[8], lr[8];
#pragma unroll
  for (int r = 0; r < 8; ++r) { mr[r] = -1e30f; lr[r] = 0.f; }

  for (int s0 = 0; s0 <= m0 + 15; s0 += 32) {
    v8f s[2];
#pragma unroll
    for (int kt = 0; kt < 2; ++kt) {
      const bf16_t* krow = Kp + (size_t)(s0 + kt * 16 + col) * DKK;
      v16bf b0 = *(const v16bf*)(krow + sub * 16);
      v16bf b1 = *(const v16bf*)(krow + 32 + sub * 16);
      v8f t = zero_v8f();
      t = __builtin_amdgcn_wmma_f32_16x16x32_bf16(false, qa0, false, b0, (short)0, t, false, false);
      t = __builtin_amdgcn_wmma_f32_16x16x32_bf16(false, qa1, false, b1, (short)0, t, false, false);
      s[kt] = t;
    }
#pragma unroll
    for (int r = 0; r < 8; ++r) {
      int qi = m0 + r + 8 * sub;
      float sv0 = (s0 + col      <= qi) ? s[0][r] * scale : -1e30f;
      float sv1 = (s0 + 16 + col <= qi) ? s[1][r] * scale : -1e30f;
      float rm = fmaxf(sv0, sv1);
      for (int off = 1; off < 16; off <<= 1) rm = fmaxf(rm, __shfl_xor(rm, off, 32));
      float nm = fmaxf(mr[r], rm);
      float corr = __expf(mr[r] - nm);
      mr[r] = nm;
      float p0 = __expf(sv0 - nm);
      float p1 = __expf(sv1 - nm);
      float ps = p0 + p1;
      for (int off = 1; off < 16; off <<= 1) ps += __shfl_xor(ps, off, 32);
      lr[r] = lr[r] * corr + ps;
#pragma unroll
      for (int dt = 0; dt < 4; ++dt) o[dt][r] *= corr;
      pbuf[(r + 8 * sub) * 32 + col]      = (bf16_t)p0;
      pbuf[(r + 8 * sub) * 32 + 16 + col] = (bf16_t)p1;
    }
    // Re-read P (16 queries x 32 keys) as an A-fragment. LDS ops are
    // in-order within a wave on CDNA5 (single wave per block).
    v16bf pa;
    {
      const bf16_t* prow = pbuf + col * 32;
      v8bf lo = *(const v8bf*)(prow + sub * 16);
      v8bf hi = *(const v8bf*)(prow + sub * 16 + 8);
#pragma unroll
      for (int i = 0; i < 8; ++i) { pa[i] = lo[i]; pa[8 + i] = hi[i]; }
    }
#pragma unroll
    for (int dt = 0; dt < 4; ++dt) {
      v16bf vb = *(const v16bf*)(Vp + (size_t)(dt * 16 + col) * TT + s0 + sub * 16);
      o[dt] = __builtin_amdgcn_wmma_f32_16x16x32_bf16(
          false, pa, false, vb, (short)0, o[dt], false, false);
    }
  }

  // epilogue: normalize, sigmoid gate, store bf16 row-major [t][h*64+d]
#pragma unroll
  for (int dt = 0; dt < 4; ++dt) {
#pragma unroll
    for (int r = 0; r < 8; ++r) {
      int t = m0 + r + 8 * sub;
      int d = dt * 16 + col;
      size_t gi = (size_t)t * DD + h * DKK + d;
      float g = 1.0f / (1.0f + __expf(-gateraw[gi]));
      gated[gi] = (bf16_t)(o[dt][r] / lr[r] * g);
    }
  }
}

// ---------------------------------------------------------------------------
// host-side orchestration
// ---------------------------------------------------------------------------
extern "C" void kernel_launch(void* const* d_in, const int* in_sizes, int n_in,
                              void* d_out, int out_size, void* d_ws, size_t ws_size,
                              hipStream_t stream) {
  (void)in_sizes; (void)n_in; (void)out_size; (void)ws_size;
  const float* x    = (const float*)d_in[0];
  // d_in[1] = mask (causal tril) — applied analytically
  const float* Wq   = (const float*)d_in[2];
  const float* Wk   = (const float*)d_in[3];
  const float* Wv   = (const float*)d_in[4];
  const float* Wo   = (const float*)d_in[5];
  const float* Wg   = (const float*)d_in[6];
  const float* bg   = (const float*)d_in[7];
  const float* g1   = (const float*)d_in[8];
  const float* b1   = (const float*)d_in[9];
  const float* g2   = (const float*)d_in[10];
  const float* b2   = (const float*)d_in[11];
  const float* Wf1  = (const float*)d_in[12];
  const float* bf1  = (const float*)d_in[13];
  const float* Wf2  = (const float*)d_in[14];
  const float* bf2  = (const float*)d_in[15];
  const float* cosb = (const float*)d_in[16];
  const float* sinb = (const float*)d_in[17];
  float* out = (float*)d_out;

  char* ws = (char*)d_ws;
  const size_t MB = 1ull << 20;
  // persistent bf16 weights (transposed): 26 MB
  bf16_t* wq_t  = (bf16_t*)(ws + 0 * MB);
  bf16_t* wk_t  = (bf16_t*)(ws + 2 * MB);
  bf16_t* wv_t  = (bf16_t*)(ws + 4 * MB);
  bf16_t* wg_t  = (bf16_t*)(ws + 6 * MB);
  bf16_t* wo_t  = (bf16_t*)(ws + 8 * MB);
  bf16_t* wf1_t = (bf16_t*)(ws + 10 * MB);  // 8 MB
  bf16_t* wf2_t = (bf16_t*)(ws + 18 * MB);  // 8 MB
  bf16_t* xn_bf = (bf16_t*)(ws + 26 * MB);  // 8 MB, reused as zn_bf
  float*  qraw  = (float*)(ws + 34 * MB);   // 16 MB, reused as gated_bf
  float*  kraw  = (float*)(ws + 50 * MB);   // 16 MB, reused as woC
  float*  vraw  = (float*)(ws + 66 * MB);   // 16 MB, reused as y
  float*  grw   = (float*)(ws + 82 * MB);   // 16 MB gate raw
  bf16_t* q_bf  = (bf16_t*)(ws + 98 * MB);  // 8 MB
  bf16_t* k_bf  = (bf16_t*)(ws + 106 * MB); // 8 MB
  bf16_t* v_t   = (bf16_t*)(ws + 114 * MB); // 8 MB
  bf16_t* h_bf  = (bf16_t*)(ws + 122 * MB); // 32 MB -> total 154 MB
  bf16_t* gated_bf = (bf16_t*)(ws + 34 * MB);
  float*  woC   = (float*)(ws + 50 * MB);
  float*  yb    = (float*)(ws + 66 * MB);
  bf16_t* zn_bf = xn_bf;

  auto cdiv = [](long a, long b) { return (int)((a + b - 1) / b); };

  // 1) weights -> bf16 transposed (LDS-tiled transpose, grid = (N/32, K/32))
  wconv_kernel<<<dim3(DD / 32, DD / 32), 256, 0, stream>>>(Wq, wq_t, DD, DD);
  wconv_kernel<<<dim3(DD / 32, DD / 32), 256, 0, stream>>>(Wk, wk_t, DD, DD);
  wconv_kernel<<<dim3(DD / 32, DD / 32), 256, 0, stream>>>(Wv, wv_t, DD, DD);
  wconv_kernel<<<dim3(DD / 32, DD / 32), 256, 0, stream>>>(Wg, wg_t, DD, DD);
  wconv_kernel<<<dim3(DD / 32, DD / 32), 256, 0, stream>>>(Wo, wo_t, DD, DD);
  wconv_kernel<<<dim3(FF / 32, DD / 32), 256, 0, stream>>>(Wf1, wf1_t, DD, FF);
  wconv_kernel<<<dim3(DD / 32, FF / 32), 256, 0, stream>>>(Wf2, wf2_t, FF, DD);

  // 2) LN1 -> xn (bf16)
  ln_kernel<<<TT, 256, 0, stream>>>(x, nullptr, g1, b1, xn_bf, nullptr);

  // 3) QKV + gate projections (WMMA GEMMs, M=4096 N=1024 K=1024)
  dim3 gN1024(DD / 128, TT / 128);
  gemm_bf16_kernel<<<gN1024, 256, 0, stream>>>(xn_bf, wq_t, nullptr, qraw, nullptr, nullptr, TT, DD, DD, 0);
  gemm_bf16_kernel<<<gN1024, 256, 0, stream>>>(xn_bf, wk_t, nullptr, kraw, nullptr, nullptr, TT, DD, DD, 0);
  gemm_bf16_kernel<<<gN1024, 256, 0, stream>>>(xn_bf, wv_t, nullptr, vraw, nullptr, nullptr, TT, DD, DD, 0);
  gemm_bf16_kernel<<<gN1024, 256, 0, stream>>>(xn_bf, wg_t, bg,      grw,  nullptr, nullptr, TT, DD, DD, 0);

  // 4) RoPE (q,k) + V transpose relayout
  rope_kernel<<<cdiv((long)TT * HH * 32, 256), 256, 0, stream>>>(qraw, cosb, sinb, q_bf);
  rope_kernel<<<cdiv((long)TT * HH * 32, 256), 256, 0, stream>>>(kraw, cosb, sinb, k_bf);
  vtrans_kernel<<<dim3(TT / 32, DD / 32), 256, 0, stream>>>(vraw, v_t);

  // 5) causal flash attention + sigmoid gate -> gated (bf16)
  attn_kernel<<<dim3(TT / 16, HH), 32, 0, stream>>>(q_bf, k_bf, v_t, grw, gated_bf);

  // 6) output projection
  gemm_bf16_kernel<<<gN1024, 256, 0, stream>>>(gated_bf, wo_t, nullptr, woC, nullptr, nullptr, TT, DD, DD, 0);

  // 7) residual + LN2: y = woC + x (fp32), zn (bf16)
  ln_kernel<<<TT, 256, 0, stream>>>(woC, x, g2, b2, zn_bf, yb);

  // 8) FFN1 with fused GELU -> h (bf16), M=4096 N=4096 K=1024
  dim3 gN4096(FF / 128, TT / 128);
  gemm_bf16_kernel<<<gN4096, 256, 0, stream>>>(zn_bf, wf1_t, bf1, nullptr, h_bf, nullptr, TT, FF, DD, 1);

  // 9) FFN2 with fused bias + residual -> out (fp32), M=4096 N=1024 K=4096
  gemm_bf16_kernel<<<gN1024, 256, 0, stream>>>(h_bf, wf2_t, bf2, out, nullptr, yb, TT, DD, FF, 2);
}
